// DeformNet_64012192579611
// MI455X (gfx1250) — matrix operand order repaired and checked by reading, not compiled, not run on hardware
//
#include <hip/hip_runtime.h>
#include <hip/hip_bf16.h>

// ---------------------------------------------------------------------------
// DeformNet on gfx1250: all per-token linears as f16 WMMA GEMMs (f32 accum).
// A-fragments built in-register from raw f32 activations (b128 loads + packed
// cvt); B-fragments pre-packed once per weight matrix. K-loop fully unrolled
// via Kt template so all loads are base+immediate (no address chains).
// ---------------------------------------------------------------------------

typedef __attribute__((ext_vector_type(16))) _Float16 v16h;
typedef __attribute__((ext_vector_type(8)))  _Float16 v8h;
typedef __attribute__((ext_vector_type(8)))  float    v8f;

#define TOK        8192      // bs * n_pts  (= bs * nv)
#define TPB        1024      // tokens per batch
#define HW_PIX     36864     // 192*192
#define MTILES     (TOK/16)  // 512 row tiles

// ---------------- gather: emb_map (b,32,HW) at choose -> (t,32) -------------
__global__ void k_gather(const float* __restrict__ emb_map,
                         const int* __restrict__ choose,
                         float* __restrict__ E32) {
  int idx = blockIdx.x * blockDim.x + threadIdx.x;
  if (idx >= TOK * 32) return;
  int t = idx >> 5, c = idx & 31;
  int b = t / TPB;
  int pix = choose[t];
  E32[(size_t)t * 32 + c] = emb_map[((size_t)b * 32 + c) * HW_PIX + pix];
}

// ---- pack A (only for C=3 inputs): X (TOK x 3) -> zero-padded fragments ----
// tile (tm,tk): 512 f16; lane l elem i:  m=l%16,  k=(i/8)*16 + (l/16)*8 + i%8
__global__ void k_packA(const float* __restrict__ X, _Float16* __restrict__ pA,
                        int C) {
  int idx = blockIdx.x * blockDim.x + threadIdx.x;
  int total = MTILES * 512;                       // Kt == 1
  if (idx >= total) return;
  int e = idx & 511, tmi = idx >> 9;
  int l = e >> 4, i = e & 15;
  int m = l & 15;
  int k = ((i >> 3) << 4) + ((l >> 4) << 3) + (i & 7);
  float v = (k < C) ? X[(size_t)(tmi * 16 + m) * C + k] : 0.0f;
  pA[idx] = (_Float16)v;
}

// ---- pack B: W (Cout x Cin row-major) -> B fragments (tiles (tn,tk)) -------
// lane l elem i:  n=l%16,  k=(i/8)*16 + (l/16)*8 + i%8   (B[k][n] = W[n][k])
__global__ void k_packB(const float* __restrict__ W, _Float16* __restrict__ pB,
                        int Cout, int Cin, int Kt) {
  int idx = blockIdx.x * blockDim.x + threadIdx.x;
  int total = (Cout / 16) * Kt * 512;
  if (idx >= total) return;
  int e = idx & 511, tile = idx >> 9;
  int tk = tile % Kt, tn = tile / Kt;
  int l = e >> 4, i = e & 15;
  int n = l & 15;
  int k = ((i >> 3) << 4) + ((l >> 4) << 3) + (i & 7);
  int kk = tk * 32 + k, nn = tn * 16 + n;
  float v = (kk < Cin) ? W[(size_t)nn * Cin + kk] : 0.0f;
  pB[idx] = (_Float16)v;
}

// ------------------------- shared GEMM epilogue -----------------------------
__device__ __forceinline__ void gemm_epilogue(
    v8f acc[4], const float* __restrict__ bias, float* __restrict__ Y,
    int lane, int tm, int tnBase, int catAdd, int ldY, int colOffY, int relu) {
  int n = lane & 15;
  int mrow = (lane >> 4) << 3;
#pragma unroll
  for (int j = 0; j < 4; ++j) {
    int tnEff = catAdd + tnBase + j;
    float bv = bias[(size_t)tnEff * 16 + n];
#pragma unroll
    for (int r = 0; r < 8; ++r) {
      float v = acc[j][r] + bv;
      if (relu) v = fmaxf(v, 0.0f);
      Y[(size_t)(tm * 16 + mrow + r) * ldY + colOffY + (tnBase + j) * 16 + n] = v;
    }
  }
}

// -------- GEMM, A built in-register from f32 activations (C % 32 == 0) -----
// one wave -> 16x64 output tile (4 N-tiles share the A fragment). Kt is a
// template constant so the K loop fully unrolls and every load is
// base + compile-time immediate offset (fits the 24-bit IOFFSET field).
template <int Kt>
__global__ void __launch_bounds__(128)
k_gemm_f32(const float* __restrict__ X, const _Float16* __restrict__ pB,
           const float* __restrict__ bias, float* __restrict__ Y,
           int ldX, int colOffX, int Nt, int ldY, int colOffY,
           int relu, const int* __restrict__ cat, int catTileOff) {
  int wid  = blockIdx.x * (blockDim.x >> 5) + (threadIdx.x >> 5);
  int lane = threadIdx.x & 31;
  int wavesPerRow = Nt >> 2;
  int tm = wid / wavesPerRow;
  if (tm >= MTILES) return;
  int tnBase = (wid % wavesPerRow) << 2;
  int catAdd = cat ? cat[(tm * 16) / TPB] * catTileOff : 0;

  v8f acc[4];
  acc[0] = (v8f){}; acc[1] = (v8f){}; acc[2] = (v8f){}; acc[3] = (v8f){};

  // A fragment: lane l -> row m = l%16, cols [kgrp, kgrp+8) and [16+kgrp, ...)
  int m = lane & 15;
  int kgrp = (lane >> 4) << 3;
  const float* xr = X + (size_t)(tm * 16 + m) * ldX + colOffX + kgrp;
  const _Float16* bPtr =
      pB + ((size_t)(catAdd + tnBase) * Kt) * 512 + lane * 16;

#pragma unroll
  for (int tk = 0; tk < Kt; ++tk) {
    v8f x0 = *(const v8f*)(xr + tk * 32);
    v8f x1 = *(const v8f*)(xr + tk * 32 + 16);
    v16h a = __builtin_shufflevector(__builtin_convertvector(x0, v8h),
                                     __builtin_convertvector(x1, v8h),
                                     0, 1, 2, 3, 4, 5, 6, 7,
                                     8, 9, 10, 11, 12, 13, 14, 15);
    v16h b0 = *(const v16h*)(bPtr + tk * 512);
    v16h b1 = *(const v16h*)(bPtr + (tk + Kt) * 512);
    v16h b2 = *(const v16h*)(bPtr + (tk + 2 * Kt) * 512);
    v16h b3 = *(const v16h*)(bPtr + (tk + 3 * Kt) * 512);
    acc[0] = __builtin_amdgcn_wmma_f32_16x16x32_f16(false, a, false, b0,
                                                    (short)0, acc[0], false, false);
    acc[1] = __builtin_amdgcn_wmma_f32_16x16x32_f16(false, a, false, b1,
                                                    (short)0, acc[1], false, false);
    acc[2] = __builtin_amdgcn_wmma_f32_16x16x32_f16(false, a, false, b2,
                                                    (short)0, acc[2], false, false);
    acc[3] = __builtin_amdgcn_wmma_f32_16x16x32_f16(false, a, false, b3,
                                                    (short)0, acc[3], false, false);
  }
  gemm_epilogue(acc, bias, Y, lane, tm, tnBase, catAdd, ldY, colOffY, relu);
}

// -------- GEMM from pre-packed A (used for the two C=3 layers, Kt=1) --------
__global__ void __launch_bounds__(128)
k_gemm_pA(const _Float16* __restrict__ pA, const _Float16* __restrict__ pB,
          const float* __restrict__ bias, float* __restrict__ Y,
          int Nt, int ldY, int colOffY, int relu) {
  int wid  = blockIdx.x * (blockDim.x >> 5) + (threadIdx.x >> 5);
  int lane = threadIdx.x & 31;
  int wavesPerRow = Nt >> 2;
  int tm = wid / wavesPerRow;
  if (tm >= MTILES) return;
  int tnBase = (wid % wavesPerRow) << 2;

  v8f acc[4];
  acc[0] = (v8f){}; acc[1] = (v8f){}; acc[2] = (v8f){}; acc[3] = (v8f){};

  v16h a = *(const v16h*)(pA + (size_t)tm * 512 + lane * 16);
  const _Float16* bp = pB + (size_t)tnBase * 512 + lane * 16;
#pragma unroll
  for (int j = 0; j < 4; ++j) {
    v16h b = *(const v16h*)(bp + (size_t)j * 512);
    acc[j] = __builtin_amdgcn_wmma_f32_16x16x32_f16(
        false, a, false, b, (short)0, acc[j], false, false);
  }
  gemm_epilogue(acc, bias, Y, lane, tm, tnBase, 0, ldY, colOffY, relu);
}

// ------------- residual broadcast add: Y[t,c] += P[t, c % Ch] ---------------
__global__ void k_add_bcast(float* __restrict__ Y, const float* __restrict__ P,
                            int C, int Ch) {
  int idx = blockIdx.x * blockDim.x + threadIdx.x;
  if (idx >= TOK * C) return;
  int t = idx / C, c = idx % C;
  Y[idx] += P[(size_t)t * Ch + (c & (Ch - 1))];
}

// --------- deform final layer: 3 selected rows of (18 x 256) per cat --------
__global__ void k_deform_out(const float* __restrict__ H256,
                             const float* __restrict__ W,
                             const float* __restrict__ bvec,
                             const int* __restrict__ cat,
                             float* __restrict__ out) {
  int idx = blockIdx.x * blockDim.x + threadIdx.x;
  if (idx >= TOK * 3) return;
  int t = idx / 3, j = idx % 3;
  int c = cat[t / TPB];
  const float* w = W + (size_t)(c * 3 + j) * 256;
  const float* x = H256 + (size_t)t * 256;
  float s = bvec[c * 3 + j];
#pragma unroll 8
  for (int k = 0; k < 256; ++k) s = fmaf(x[k], w[k], s);
  out[idx] = s;
}

// ---------------------------------------------------------------------------
extern "C" void kernel_launch(void* const* d_in, const int* in_sizes, int n_in,
                              void* d_out, int out_size, void* d_ws, size_t ws_size,
                              hipStream_t stream) {
  const float* points  = (const float*)d_in[0];
  const float* emb_map = (const float*)d_in[1];
  const int*   choose  = (const int*)  d_in[2];
  const int*   cat_id  = (const int*)  d_in[3];
  const float* prior   = (const float*)d_in[4];
  auto P = [&](int i) { return (const float*)d_in[i]; };
  // param indices (dict order, W then b per layer):
  // 5/6 7/8 9/10: ig0..2 | 11/12: ic | 13/14: iglob | 15..20: cl0..2
  // 21/22: cg | 23..28: assign0..2 | 29..34: deform0..2 | 35..38: t64 | 39..42: t128

  // ----- workspace bump layout (bytes) -----
  char* ws = (char*)d_ws;
  const size_t MB = 1u << 20;
  float*    E32    = (float*)(ws + 0);          //  1 MB  (TOK x 32)
  float*    CAT128 = (float*)(ws + 1 * MB);     //  4 MB  (TOK x 128) [pts | emb]
  float*    TMPA   = (float*)(ws + 5 * MB);     //  2 MB  (TOK x 64)
  float*    TMPB   = (float*)(ws + 7 * MB);     //  2 MB  (TOK x 64)
  float*    H512   = (float*)(ws + 9 * MB);     // 16 MB  (TOK x 512)
  float*    H256   = (float*)(ws + 25 * MB);    //  8 MB  (TOK x 256)
  float*    CAT256 = (float*)(ws + 33 * MB);    //  8 MB  (TOK x 256) [inst_g | cat_g]
  float*    PQ     = (float*)(ws + 41 * MB);    //  4 MB  (p: TOKx64 / q: TOKx128)
  _Float16* pA     = (_Float16*)(ws + 45 * MB); //  1 MB  (TOK x 32 f16, C=3 pad)
  _Float16* pB     = (_Float16*)(ws + 46 * MB); //  4 MB  (max 6144 x 256 f16)

  auto packB = [&](const float* W, int Cout, int Cin, int Kt) {
    int total = (Cout / 16) * Kt * 512;
    k_packB<<<(total + 255) / 256, 256, 0, stream>>>(W, pB, Cout, Cin, Kt);
  };
  auto gemmF = [&](const float* X, int ldX, int colOffX, int Kt,
                   const float* bias, float* Y, int Nt, int ldY, int colOffY,
                   int relu, const int* cat, int catTileOff) {
    int waves = MTILES * (Nt >> 2);
    int blocks = (waves + 3) / 4;
    switch (Kt) {
      case 1:
        k_gemm_f32<1><<<blocks, 128, 0, stream>>>(X, pB, bias, Y, ldX, colOffX,
                                                  Nt, ldY, colOffY, relu, cat,
                                                  catTileOff);
        break;
      case 2:
        k_gemm_f32<2><<<blocks, 128, 0, stream>>>(X, pB, bias, Y, ldX, colOffX,
                                                  Nt, ldY, colOffY, relu, cat,
                                                  catTileOff);
        break;
      case 4:
        k_gemm_f32<4><<<blocks, 128, 0, stream>>>(X, pB, bias, Y, ldX, colOffX,
                                                  Nt, ldY, colOffY, relu, cat,
                                                  catTileOff);
        break;
      case 8:
        k_gemm_f32<8><<<blocks, 128, 0, stream>>>(X, pB, bias, Y, ldX, colOffX,
                                                  Nt, ldY, colOffY, relu, cat,
                                                  catTileOff);
        break;
      default:
        k_gemm_f32<16><<<blocks, 128, 0, stream>>>(X, pB, bias, Y, ldX, colOffX,
                                                   Nt, ldY, colOffY, relu, cat,
                                                   catTileOff);
        break;
    }
  };
  auto gemmP = [&](const float* bias, float* Y, int Nt, int ldY, int colOffY,
                   int relu) {
    int waves = MTILES * (Nt >> 2);
    k_gemm_pA<<<(waves + 3) / 4, 128, 0, stream>>>(pA, pB, bias, Y, Nt, ldY,
                                                   colOffY, relu);
  };

  // 1) gather image embeddings, then ic: 32 -> 64 relu  -> CAT128[:,64:128]
  k_gather<<<(TOK * 32 + 255) / 256, 256, 0, stream>>>(emb_map, choose, E32);
  packB(P(11), 64, 32, 1);
  gemmF(E32, 32, 0, 1, P(12), CAT128, 4, 128, 64, 1, nullptr, 0);

  // 2) instance geometry MLP: 3 -> 64 -> 64 -> 64  -> CAT128[:,0:64]
  k_packA<<<(MTILES * 512 + 255) / 256, 256, 0, stream>>>(points, pA, 3);
  packB(P(5), 64, 3, 1);
  gemmP(P(6), TMPA, 4, 64, 0, 1);
  packB(P(7), 64, 64, 2);
  gemmF(TMPA, 64, 0, 2, P(8), TMPB, 4, 64, 0, 1, nullptr, 0);
  packB(P(9), 64, 64, 2);
  gemmF(TMPB, 64, 0, 2, P(10), CAT128, 4, 128, 0, 1, nullptr, 0);

  // 3) relation t64: relu(concat) is identity (both inputs post-ReLU)
  packB(P(35), 512, 128, 4);
  gemmF(CAT128, 128, 0, 4, P(36), H512, 32, 512, 0, 1, nullptr, 0);
  packB(P(37), 64, 512, 16);
  gemmF(H512, 512, 0, 16, P(38), PQ, 4, 64, 0, 0, nullptr, 0);
  k_add_bcast<<<(TOK * 128 + 255) / 256, 256, 0, stream>>>(CAT128, PQ, 128, 64);

  // 4) inst_global = relu(lin128->128)  -> CAT256[:,0:128]
  packB(P(13), 128, 128, 4);
  gemmF(CAT128, 128, 0, 4, P(14), CAT256, 8, 256, 0, 1, nullptr, 0);

  // 5) category branch: 3 -> 64 -> 64 -> 64, cg 64 -> 128 -> CAT256[:,128:256]
  k_packA<<<(MTILES * 512 + 255) / 256, 256, 0, stream>>>(prior, pA, 3);
  packB(P(15), 64, 3, 1);
  gemmP(P(16), TMPA, 4, 64, 0, 1);
  packB(P(17), 64, 64, 2);
  gemmF(TMPA, 64, 0, 2, P(18), TMPB, 4, 64, 0, 1, nullptr, 0);
  packB(P(19), 64, 64, 2);
  gemmF(TMPB, 64, 0, 2, P(20), TMPA, 4, 64, 0, 1, nullptr, 0);
  packB(P(21), 128, 64, 2);
  gemmF(TMPA, 64, 0, 2, P(22), CAT256, 8, 256, 128, 1, nullptr, 0);

  // 6) relation t128 on concat (identity relu), residual into both halves
  packB(P(39), 512, 256, 8);
  gemmF(CAT256, 256, 0, 8, P(40), H512, 32, 512, 0, 1, nullptr, 0);
  packB(P(41), 128, 512, 16);
  gemmF(H512, 512, 0, 16, P(42), PQ, 8, 128, 0, 0, nullptr, 0);
  k_add_bcast<<<(TOK * 256 + 255) / 256, 256, 0, stream>>>(CAT256, PQ, 256, 128);

  // 7) assignment head: 128 -> 512 -> 256 -> 1024 (category-selected rows)
  packB(P(23), 512, 128, 4);
  gemmF(CAT256, 256, 0, 4, P(24), H512, 32, 512, 0, 1, nullptr, 0);
  packB(P(25), 256, 512, 16);
  gemmF(H512, 512, 0, 16, P(26), H256, 16, 256, 0, 1, nullptr, 0);
  packB(P(27), 6144, 256, 8);
  gemmF(H256, 256, 0, 8, P(28), (float*)d_out, 64, 1024, 0, 0, cat_id, 64);

  // 8) deformation head: 128 -> 512 -> 256 -> 3 (selected rows, VALU finish)
  packB(P(29), 512, 128, 4);
  gemmF(CAT256, 256, 128, 4, P(30), H512, 32, 512, 0, 1, nullptr, 0);
  packB(P(31), 256, 512, 16);
  gemmF(H512, 512, 0, 16, P(32), H256, 16, 256, 0, 1, nullptr, 0);
  k_deform_out<<<(TOK * 3 + 255) / 256, 256, 0, stream>>>(
      H256, P(33), P(34), cat_id, (float*)d_out + (size_t)TOK * 1024);

  (void)in_sizes; (void)n_in; (void)out_size; (void)ws_size;
}
